// DLSTM_python_Feature_Net_78958678769797
// MI455X (gfx1250) — compile-verified, matrix operand order checked by reading, not run on hardware
//
#include <hip/hip_runtime.h>
#include <hip/hip_bf16.h>
#include <math.h>

// ---------------------------------------------------------------------------
// Problem dims (match reference)
// ---------------------------------------------------------------------------
#define NB   64       // batch
#define NT   512      // sequence length
#define NE   128      // embedding dim
#define NH   256      // hidden
#define G4H  1024     // 4*NH (gates)
#define KF   131072   // NT * 2H / POOL  : classifier reduction dim
#define NOUT 256
#define KSPLIT 32     // K-splits in classifier GEMM

typedef __attribute__((ext_vector_type(16))) __bf16 v16bf;
typedef __attribute__((ext_vector_type(8)))  __bf16 v8bf;
typedef __attribute__((ext_vector_type(8)))  float  v8f;

// ---------------------------------------------------------------------------
// WMMA bf16 fragment loader.
// ISA layout (16-bit A 16x32 / B 32x16): lane L (<16) holds K = {0..7, 16..23},
// lane L (>=16) holds K = {8..15, 24..31}; caller passes the row/col base ptr
// already offset by kb = 32*ks + 8*(L>>4).  Two b128 loads.
// ---------------------------------------------------------------------------
static __device__ __forceinline__ v16bf load_frag(const __bf16* p) {
  v8bf lo = *(const v8bf*)(p);
  v8bf hi = *(const v8bf*)(p + 16);
  v16bf f;
#pragma unroll
  for (int i = 0; i < 8; ++i) { f[i] = lo[i]; f[i + 8] = hi[i]; }
  return f;
}

static __device__ __forceinline__ v8f v8f_zero() {
  v8f z = {0.f, 0.f, 0.f, 0.f, 0.f, 0.f, 0.f, 0.f};
  return z;
}

static __device__ __forceinline__ float sigf(float x) {
  return 1.0f / (1.0f + __expf(-x));
}

// ---------------------------------------------------------------------------
// K0a: embedding gather fp32 -> bf16   xe[b*T+t][e] = emb[x[b][t]][e]
// ---------------------------------------------------------------------------
__global__ void embed_gather_kernel(const int* __restrict__ x,
                                    const float* __restrict__ emb,
                                    __bf16* __restrict__ xe) {
  const int bt = blockIdx.x;            // 0 .. NB*NT-1
  const int e  = threadIdx.x;           // 0 .. 127
  const int row = x[bt];
  xe[(size_t)bt * NE + e] = (__bf16)emb[(size_t)row * NE + e];
}

// ---------------------------------------------------------------------------
// K0b: generic fp32 -> bf16 convert
// ---------------------------------------------------------------------------
__global__ void cvt_bf16_kernel(const float* __restrict__ src,
                                __bf16* __restrict__ dst, int n) {
  int i = blockIdx.x * 256 + threadIdx.x;
  if (i < n) dst[i] = (__bf16)src[i];
}

// ---------------------------------------------------------------------------
// K0c: fused bias  bias = bih + bhh  (both directions)
// ---------------------------------------------------------------------------
__global__ void bias_sum_kernel(const float* bihF, const float* bhhF,
                                const float* bihB, const float* bhhB,
                                float* biasF, float* biasB) {
  int i = blockIdx.x * 256 + threadIdx.x;
  if (i < G4H) {
    biasF[i] = bihF[i] + bhhF[i];
    biasB[i] = bihB[i] + bhhB[i];
  }
}

// ---------------------------------------------------------------------------
// K1: input projection  xg[dir] = xe @ Wih[dir]^T + bias[dir]
//     M = NB*NT = 32768 (grid.x = 2048 tiles), N = 1024, K = 128 (4 k-steps)
//     256 threads = 8 waves; wave w owns N-tiles 8w..8w+7 (A frag reused 8x)
// ---------------------------------------------------------------------------
__global__ __launch_bounds__(256)
void input_proj_kernel(const __bf16* __restrict__ xe,
                       const __bf16* __restrict__ WihF,
                       const __bf16* __restrict__ WihB,
                       const float* __restrict__ biasF,
                       const float* __restrict__ biasB,
                       float* __restrict__ xgF,
                       float* __restrict__ xgB) {
  const int dir = blockIdx.y;
  const __bf16* Wih  = dir ? WihB  : WihF;
  const float*  bias = dir ? biasB : biasF;
  float*        xg   = dir ? xgB   : xgF;

  const int mb   = blockIdx.x;          // M tile
  const int tid  = threadIdx.x;
  const int lane = tid & 31;
  const int wave = tid >> 5;
  const int half = lane >> 4;
  const int l16  = lane & 15;

  v8f acc[8];
#pragma unroll
  for (int i = 0; i < 8; ++i) acc[i] = v8f_zero();

  const int m = mb * 16 + l16;
#pragma unroll
  for (int ks = 0; ks < 4; ++ks) {
    const int kb = ks * 32 + half * 8;
    v16bf a = load_frag(xe + (size_t)m * NE + kb);
#pragma unroll
    for (int ni = 0; ni < 8; ++ni) {
      const int n = (wave * 8 + ni) * 16 + l16;      // B col = Wih row
      v16bf b = load_frag(Wih + (size_t)n * NE + kb);
      acc[ni] = __builtin_amdgcn_wmma_f32_16x16x32_bf16(
          false, a, false, b, (short)0, acc[ni], false, false);
    }
  }
#pragma unroll
  for (int ni = 0; ni < 8; ++ni) {
    const int n = (wave * 8 + ni) * 16 + l16;
    const float bv = bias[n];
#pragma unroll
    for (int r = 0; r < 8; ++r) {
      const int mr = mb * 16 + r + half * 8;
      xg[(size_t)mr * G4H + n] = acc[ni][r] + bv;
    }
  }
}

// ---------------------------------------------------------------------------
// K2: persistent bidirectional LSTM recurrence. grid = 2 (dir), 1024 threads.
//     Gate-aligned tiling: wave w owns j-tile jt = w&15 and M-tiles
//     {2*(w>>4), 2*(w>>4)+1}, computing accumulator tiles (mt, q*16+jt) for
//     all four gates q. The full cell update then happens in registers:
//     i/f/g/o for a given (batch m, hidden j) sit in the same lane across
//     the 4 gate accumulators. h -> LDS (bf16, next step's WMMA A operand);
//     c register-resident; max-pool via wave32 shfl_xor on adjacent lanes.
// ---------------------------------------------------------------------------
__global__ __launch_bounds__(1024)
void lstm_recurrent_kernel(const __bf16* __restrict__ WhhF,
                           const __bf16* __restrict__ WhhB,
                           const float* __restrict__ xgF,
                           const float* __restrict__ xgB,
                           __bf16* __restrict__ pooled) { // [NB][KF]
  const int dir = blockIdx.x;
  const __bf16* Whh = dir ? WhhB : WhhF;
  const float*  xg  = dir ? xgB  : xgF;

  __shared__ __bf16 h_lds[NB * NH];     // 32 KB

  const int tid  = threadIdx.x;
  const int lane = tid & 31;
  const int wave = tid >> 5;            // 0..31
  const int half = lane >> 4;
  const int l16  = lane & 15;
  const int jt   = wave & 15;           // hidden column tile (0..15)
  const int mh   = wave >> 4;           // M-tile pair {2mh, 2mh+1}
  const int j    = jt * 16 + l16;       // hidden index owned by this lane

  // zero h
  for (int i = tid; i < NB * NH; i += 1024) h_lds[i] = (__bf16)0.0f;

  // cell state: lane owns c for (m = (2mh+mt)*16 + r + 8*half, j)
  float c[2][8];
#pragma unroll
  for (int mt = 0; mt < 2; ++mt)
#pragma unroll
    for (int r = 0; r < 8; ++r) c[mt][r] = 0.0f;

  for (int step = 0; step < NT; ++step) {
    const int t = dir ? (NT - 1 - step) : step;
    __syncthreads();                    // h from previous step visible

    // ---- GEMM: acc[mt][q] = h @ Whh^T tile (mt, q*16+jt) ----
    v8f acc[2][4];
#pragma unroll
    for (int mt = 0; mt < 2; ++mt)
#pragma unroll
      for (int q = 0; q < 4; ++q) acc[mt][q] = v8f_zero();

#pragma unroll
    for (int ks = 0; ks < 8; ++ks) {
      const int kb = ks * 32 + half * 8;
      v16bf bfr[4];
#pragma unroll
      for (int q = 0; q < 4; ++q) {
        const int n = q * NH + j;                    // B col = Whh row
        bfr[q] = load_frag(Whh + (size_t)n * NH + kb);
      }
#pragma unroll
      for (int mt = 0; mt < 2; ++mt) {
        const int m = (mh * 2 + mt) * 16 + l16;
        v16bf a = load_frag(h_lds + m * NH + kb);
#pragma unroll
        for (int q = 0; q < 4; ++q) {
          acc[mt][q] = __builtin_amdgcn_wmma_f32_16x16x32_bf16(
              false, a, false, bfr[q], (short)0, acc[mt][q], false, false);
        }
      }
    }
    __syncthreads();                    // all waves done reading h_lds

    // ---- register-resident gates + cell update + pool ----
#pragma unroll
    for (int mt = 0; mt < 2; ++mt) {
#pragma unroll
      for (int r = 0; r < 8; ++r) {
        const int m = (mh * 2 + mt) * 16 + r + half * 8;   // batch row
        const float* xr = xg + ((size_t)m * NT + t) * G4H + j;
        const float ig = sigf (acc[mt][0][r] + xr[0 * NH]);
        const float fg = sigf (acc[mt][1][r] + xr[1 * NH]);
        const float gg = tanhf(acc[mt][2][r] + xr[2 * NH]);
        const float og = sigf (acc[mt][3][r] + xr[3 * NH]);
        c[mt][r] = fg * c[mt][r] + ig * gg;
        const float h = og * tanhf(c[mt][r]);
        h_lds[m * NH + j] = (__bf16)h;
        // MaxPool1d(k=2): adjacent j live in adjacent lanes
        const float ho = __shfl_xor(h, 1, 32);
        if ((l16 & 1) == 0) {
          pooled[(size_t)m * KF + (size_t)t * 256 + dir * 128 + (j >> 1)] =
              (__bf16)fmaxf(h, ho);
        }
      }
    }
  }
}

// ---------------------------------------------------------------------------
// K3: classifier GEMM  slab[ks] = pooled[64xKchunk] @ W1chunk^T  (split-K)
//     grid = (16 N-tiles, KSPLIT); one wave per block, 4 M-tiles per wave.
// ---------------------------------------------------------------------------
__global__ __launch_bounds__(32)
void final_gemm_kernel(const __bf16* __restrict__ pooled,  // [64][KF]
                       const __bf16* __restrict__ W1b,     // [256][KF]
                       float* __restrict__ slab) {         // [KSPLIT][64][256]
  const int nt     = blockIdx.x;       // 0..15
  const int ksplit = blockIdx.y;       // 0..KSPLIT-1
  const int lane = threadIdx.x;
  const int half = lane >> 4;
  const int l16  = lane & 15;

  v8f acc[4];
#pragma unroll
  for (int mt = 0; mt < 4; ++mt) acc[mt] = v8f_zero();

  const int n = nt * 16 + l16;
  const __bf16* wrow = W1b + (size_t)n * KF;
  const int kstep0 = ksplit * (KF / 32 / KSPLIT);   // 128 k-steps per split

  for (int ks = 0; ks < (KF / 32 / KSPLIT); ++ks) {
    const int kb = (kstep0 + ks) * 32 + half * 8;
    __builtin_prefetch((const void*)(wrow + kb + 1024), 0, 0);  // stream W1
    v16bf b = load_frag(wrow + kb);
#pragma unroll
    for (int mt = 0; mt < 4; ++mt) {
      v16bf a = load_frag(pooled + (size_t)(mt * 16 + l16) * KF + kb);
      acc[mt] = __builtin_amdgcn_wmma_f32_16x16x32_bf16(
          false, a, false, b, (short)0, acc[mt], false, false);
    }
  }
  float* sl = slab + (size_t)ksplit * NB * NOUT;
#pragma unroll
  for (int mt = 0; mt < 4; ++mt) {
#pragma unroll
    for (int r = 0; r < 8; ++r) {
      const int m = mt * 16 + r + half * 8;
      sl[(size_t)m * NOUT + nt * 16 + l16] = acc[mt][r];
    }
  }
}

// ---------------------------------------------------------------------------
// K4: deterministic slab reduction + bias + ReLU
// ---------------------------------------------------------------------------
__global__ void reduce_bias_relu_kernel(const float* __restrict__ slab,
                                        const float* __restrict__ b1,
                                        float* __restrict__ out) {
  int idx = blockIdx.x * 256 + threadIdx.x;    // 0 .. NB*NOUT-1
  if (idx >= NB * NOUT) return;
  float s = b1[idx & (NOUT - 1)];
#pragma unroll 4
  for (int k = 0; k < KSPLIT; ++k) s += slab[(size_t)k * NB * NOUT + idx];
  out[idx] = fmaxf(s, 0.0f);
}

// ---------------------------------------------------------------------------
// Launch
// ---------------------------------------------------------------------------
extern "C" void kernel_launch(void* const* d_in, const int* in_sizes, int n_in,
                              void* d_out, int out_size, void* d_ws, size_t ws_size,
                              hipStream_t stream) {
  const int*   x     = (const int*)  d_in[0];
  const float* emb   = (const float*)d_in[1];
  const float* Wih_f = (const float*)d_in[2];
  const float* Whh_f = (const float*)d_in[3];
  const float* bih_f = (const float*)d_in[4];
  const float* bhh_f = (const float*)d_in[5];
  const float* Wih_b = (const float*)d_in[6];
  const float* Whh_b = (const float*)d_in[7];
  const float* bih_b = (const float*)d_in[8];
  const float* bhh_b = (const float*)d_in[9];
  const float* W1    = (const float*)d_in[10];
  const float* b1    = (const float*)d_in[11];
  float* out = (float*)d_out;

  char* ws = (char*)d_ws;
  size_t off = 0;
  auto alloc = [&](size_t bytes) -> void* {
    void* p = ws + off;
    off += (bytes + 255) & ~(size_t)255;
    return p;
  };

  __bf16* xe_bf   = (__bf16*)alloc((size_t)NB * NT * NE * 2);
  __bf16* WihF_bf = (__bf16*)alloc((size_t)G4H * NE * 2);
  __bf16* WihB_bf = (__bf16*)alloc((size_t)G4H * NE * 2);
  __bf16* WhhF_bf = (__bf16*)alloc((size_t)G4H * NH * 2);
  __bf16* WhhB_bf = (__bf16*)alloc((size_t)G4H * NH * 2);
  __bf16* W1_bf   = (__bf16*)alloc((size_t)NOUT * KF * 2);
  float*  biasF   = (float*) alloc((size_t)G4H * 4);
  float*  biasB   = (float*) alloc((size_t)G4H * 4);
  float*  xgF     = (float*) alloc((size_t)NB * NT * G4H * 4);
  float*  xgB     = (float*) alloc((size_t)NB * NT * G4H * 4);
  __bf16* pooled  = (__bf16*)alloc((size_t)NB * KF * 2);
  float*  slab    = (float*) alloc((size_t)KSPLIT * NB * NOUT * 4);

  // ---- stage 0: gather + precision conversion (independent, all on stream)
  embed_gather_kernel<<<NB * NT, NE, 0, stream>>>(x, emb, xe_bf);
  cvt_bf16_kernel<<<(G4H * NE + 255) / 256, 256, 0, stream>>>(Wih_f, WihF_bf, G4H * NE);
  cvt_bf16_kernel<<<(G4H * NE + 255) / 256, 256, 0, stream>>>(Wih_b, WihB_bf, G4H * NE);
  cvt_bf16_kernel<<<(G4H * NH + 255) / 256, 256, 0, stream>>>(Whh_f, WhhF_bf, G4H * NH);
  cvt_bf16_kernel<<<(G4H * NH + 255) / 256, 256, 0, stream>>>(Whh_b, WhhB_bf, G4H * NH);
  cvt_bf16_kernel<<<(NOUT * KF + 255) / 256, 256, 0, stream>>>(W1, W1_bf, NOUT * KF);
  bias_sum_kernel<<<(G4H + 255) / 256, 256, 0, stream>>>(bih_f, bhh_f, bih_b, bhh_b,
                                                         biasF, biasB);

  // ---- stage 1: batched input projection (both directions)
  input_proj_kernel<<<dim3(NB * NT / 16, 2), 256, 0, stream>>>(
      xe_bf, WihF_bf, WihB_bf, biasF, biasB, xgF, xgB);

  // ---- stage 2: serial recurrence, one persistent workgroup per direction
  lstm_recurrent_kernel<<<2, 1024, 0, stream>>>(WhhF_bf, WhhB_bf, xgF, xgB, pooled);

  // ---- stage 3: classifier (split-K) + deterministic reduction
  final_gemm_kernel<<<dim3(NOUT / 16, KSPLIT), 32, 0, stream>>>(pooled, W1_bf, slab);
  reduce_bias_relu_kernel<<<(NB * NOUT + 255) / 256, 256, 0, stream>>>(slab, b1, out);
}
